// WeightedSharedExpertMLP_51342039056413
// MI455X (gfx1250) — matrix operand order repaired and checked by reading, not compiled, not run on hardware
//
#include <hip/hip_runtime.h>
#include <math.h>

typedef __bf16 bf16;
typedef __attribute__((ext_vector_type(16))) __bf16 v16bf;
typedef __attribute__((ext_vector_type(8)))  __bf16 v8bf;
typedef __attribute__((ext_vector_type(2)))  __bf16 v2bf;
typedef __attribute__((ext_vector_type(8)))  float  v8f;

#define N_ROWS 16384
#define D_IN   512
#define D_HID  2048
#define D_OUT  512
#define K1     576    // 512 + 4*16 LoRA columns folded into K
#define K2     2112   // 2048 + 4*16
#define SCALING 2.0f

#define EPI_BIAS 0
#define EPI_GELU 1
#define EPI_LORA 2

// Build a 16-element bf16 WMMA fragment from two 16-byte chunks.
__device__ __forceinline__ v16bf load_frag16(const bf16* p0, const bf16* p1) {
  v8bf lo = *(const v8bf*)p0;
  v8bf hi = *(const v8bf*)p1;
  v16bf r;
#pragma unroll
  for (int i = 0; i < 8; ++i) { r[i] = lo[i]; r[i + 8] = hi[i]; }
  return r;
}

__device__ __forceinline__ v8f wmma_bf16(v16bf a, v16bf b, v8f c) {
  return __builtin_amdgcn_wmma_f32_16x16x32_bf16(false, a, false, b, (short)0, c, false, false);
}

// ---------------- xa[:, 0:512] = bf16(x) --------------------------------
__global__ __launch_bounds__(256) void convert_x_kernel(
    const float* __restrict__ x, bf16* __restrict__ xa) {
  int t  = blockIdx.x * 256 + threadIdx.x;   // one thread per 2 floats
  int n  = t >> 8;
  int cp = t & 255;
  const float* p = x + (size_t)n * D_IN + 2 * cp;
  v2bf v;
  v[0] = (bf16)p[0];
  v[1] = (bf16)p[1];
  *(v2bf*)(xa + (size_t)n * K1 + 2 * cp) = v;
}

// ---------------- W1aT [2048, 576] bf16 (transposed, LoRA-B appended) ---
__global__ __launch_bounds__(256) void pack_w1_kernel(
    const float* __restrict__ W1, const float* __restrict__ B1,
    bf16* __restrict__ W1aT) {
  int idx = blockIdx.x * 256 + threadIdx.x;
  if (idx >= D_HID * K1) return;
  int k = idx % K1, n = idx / K1;
  float v = (k < D_IN) ? W1[(size_t)k * D_HID + n]
                       : B1[(size_t)(k - D_IN) * D_HID + n];  // [64,2048] flat
  W1aT[(size_t)n * K1 + k] = (bf16)v;
}

// ---------------- W2aT [512, 2112] bf16 ---------------------------------
__global__ __launch_bounds__(256) void pack_w2_kernel(
    const float* __restrict__ W2, const float* __restrict__ B2,
    bf16* __restrict__ W2aT) {
  int idx = blockIdx.x * 256 + threadIdx.x;
  if (idx >= D_OUT * K2) return;
  int k = idx % K2, n = idx / K2;
  float v = (k < D_HID) ? W2[(size_t)k * D_OUT + n]
                        : B2[(size_t)(k - D_HID) * D_OUT + n]; // [64,512] flat
  W2aT[(size_t)n * K2 + k] = (bf16)v;
}

// -------- A1T [64, 512] bf16: A1T[16b+r, i] = lora_fc1_A[b, i, r] -------
__global__ __launch_bounds__(256) void pack_a1_kernel(
    const float* __restrict__ A1, bf16* __restrict__ A1T) {
  int idx = blockIdx.x * 256 + threadIdx.x;      // 64*512
  int i = idx & (D_IN - 1);
  int c = idx >> 9;
  int b = c >> 4, r = c & 15;
  A1T[idx] = (bf16)A1[((size_t)b * D_IN + i) * 16 + r];
}

// -------- A2T [64, 2048] bf16 -------------------------------------------
__global__ __launch_bounds__(256) void pack_a2_kernel(
    const float* __restrict__ A2, bf16* __restrict__ A2T) {
  int idx = blockIdx.x * 256 + threadIdx.x;      // 64*2048
  int i = idx & (D_HID - 1);
  int c = idx >> 11;
  int b = c >> 4, r = c & 15;
  A2T[idx] = (bf16)A2[((size_t)b * D_HID + i) * 16 + r];
}

// ---------------- WMMA GEMM: C = A[:, :KLOOP] * BT^T (+epilogue) --------
// Block = 8 waves sharing one 64-col B strip (B chunks hit WGP$ 7/8 times).
// Each wave owns a 32Mx64N tile: 2 A frags x 4 B frags -> 8 WMMA per k-step
// (1.5 b128 loads per WMMA), software double-buffered K loop.
template<int KLOOP, int LDA, int EPI, typename OutT>
__global__ __launch_bounds__(256) void gemm_wmma_kernel(
    const bf16* __restrict__ A,        // [M, LDA] row-major
    const bf16* __restrict__ BT,       // [N, KLOOP] row-major (transposed W)
    const float* __restrict__ aux,     // bias[N] (EPI_BIAS/GELU) or bw[M,4] (EPI_LORA)
    OutT* __restrict__ C, int ldc,
    int n_strips) {                    // Ncols/64
  int wv   = threadIdx.x >> 5;
  int lane = threadIdx.x & 31;
  int bm   = blockIdx.x / n_strips;
  int bn   = blockIdx.x % n_strips;
  int m0   = (bm * 8 + wv) * 32;
  int n0   = bn * 64;
  int half = lane >> 4;   // K-half selector within fragments
  int l15  = lane & 15;

  // A: lane row m0+l15; elems 0..7 <- k0+half*8, elems 8..15 <- k0+16+half*8
  const bf16* arow0 = A + (size_t)(m0 + l15) * LDA + half * 8;
  const bf16* arow1 = arow0 + (size_t)16 * LDA;
  // B: lane col n0+j*16+l15; elems 0..7 <- k0+half*8, elems 8..15 <- k0+16+half*8
  const bf16* bbase = BT + (size_t)(n0 + l15) * KLOOP + half * 8;
  const size_t bstep = (size_t)16 * KLOOP;

  v8f c00 = {}, c01 = {}, c02 = {}, c03 = {};
  v8f c10 = {}, c11 = {}, c12 = {}, c13 = {};

  v16bf a0 = load_frag16(arow0, arow0 + 16);
  v16bf a1 = load_frag16(arow1, arow1 + 16);
  v16bf b0 = load_frag16(bbase + 0 * bstep, bbase + 0 * bstep + 16);
  v16bf b1 = load_frag16(bbase + 1 * bstep, bbase + 1 * bstep + 16);
  v16bf b2 = load_frag16(bbase + 2 * bstep, bbase + 2 * bstep + 16);
  v16bf b3 = load_frag16(bbase + 3 * bstep, bbase + 3 * bstep + 16);

  for (int k0 = 0; k0 < KLOOP - 32; k0 += 32) {
    int kn = k0 + 32;
    v16bf a0n = load_frag16(arow0 + kn, arow0 + kn + 16);
    v16bf a1n = load_frag16(arow1 + kn, arow1 + kn + 16);
    v16bf b0n = load_frag16(bbase + 0 * bstep + kn, bbase + 0 * bstep + kn + 16);
    v16bf b1n = load_frag16(bbase + 1 * bstep + kn, bbase + 1 * bstep + kn + 16);
    v16bf b2n = load_frag16(bbase + 2 * bstep + kn, bbase + 2 * bstep + kn + 16);
    v16bf b3n = load_frag16(bbase + 3 * bstep + kn, bbase + 3 * bstep + kn + 16);
    c00 = wmma_bf16(a0, b0, c00);
    c01 = wmma_bf16(a0, b1, c01);
    c02 = wmma_bf16(a0, b2, c02);
    c03 = wmma_bf16(a0, b3, c03);
    c10 = wmma_bf16(a1, b0, c10);
    c11 = wmma_bf16(a1, b1, c11);
    c12 = wmma_bf16(a1, b2, c12);
    c13 = wmma_bf16(a1, b3, c13);
    a0 = a0n; a1 = a1n; b0 = b0n; b1 = b1n; b2 = b2n; b3 = b3n;
  }
  c00 = wmma_bf16(a0, b0, c00);
  c01 = wmma_bf16(a0, b1, c01);
  c02 = wmma_bf16(a0, b2, c02);
  c03 = wmma_bf16(a0, b3, c03);
  c10 = wmma_bf16(a1, b0, c10);
  c11 = wmma_bf16(a1, b1, c11);
  c12 = wmma_bf16(a1, b2, c12);
  c13 = wmma_bf16(a1, b3, c13);

  // Epilogue: acc element i = row m0 + t*16 + half*8 + i, col n0 + j*16 + l15
#pragma unroll
  for (int t = 0; t < 2; ++t) {
#pragma unroll
    for (int j = 0; j < 4; ++j) {
      v8f a;
      if (t == 0) a = (j == 0) ? c00 : (j == 1) ? c01 : (j == 2) ? c02 : c03;
      else        a = (j == 0) ? c10 : (j == 1) ? c11 : (j == 2) ? c12 : c13;
      int n = n0 + j * 16 + l15;
      float bs = (EPI == EPI_LORA) ? 0.f : aux[n];
#pragma unroll
      for (int i = 0; i < 8; ++i) {
        int row = m0 + t * 16 + half * 8 + i;
        float v = a[i];
        if (EPI == EPI_LORA) {
          v *= SCALING * aux[row * 4 + j];   // n0==0, band = j
        } else {
          v += bs;
        }
        if (EPI == EPI_GELU) v = 0.5f * v * (1.0f + erff(v * 0.70710678118654752f));
        C[(size_t)row * ldc + n] = (OutT)v;
      }
    }
  }
}

extern "C" void kernel_launch(void* const* d_in, const int* in_sizes, int n_in,
                              void* d_out, int out_size, void* d_ws, size_t ws_size,
                              hipStream_t stream) {
  const float* x  = (const float*)d_in[0];
  const float* bw = (const float*)d_in[1];
  const float* W1 = (const float*)d_in[2];
  const float* b1 = (const float*)d_in[3];
  const float* W2 = (const float*)d_in[4];
  const float* b2 = (const float*)d_in[5];
  const float* A1 = (const float*)d_in[6];
  const float* B1 = (const float*)d_in[7];
  const float* A2 = (const float*)d_in[8];
  const float* B2 = (const float*)d_in[9];
  float* out = (float*)d_out;

  // Workspace layout (bf16), ~93 MB, 16B-aligned sections.
  bf16* xa  = (bf16*)d_ws;                       // [16384, 576]
  bf16* ha  = xa  + (size_t)N_ROWS * K1;         // [16384, 2112]
  bf16* w1t = ha  + (size_t)N_ROWS * K2;         // [2048, 576]
  bf16* w2t = w1t + (size_t)D_HID * K1;          // [512, 2112]
  bf16* a1t = w2t + (size_t)D_OUT * K2;          // [64, 512]
  bf16* a2t = a1t + (size_t)64 * D_IN;           // [64, 2048]

  // Stage 0: packing / conversion (tiny)
  convert_x_kernel<<<N_ROWS, 256, 0, stream>>>(x, xa);
  pack_w1_kernel<<<(D_HID * K1 + 255) / 256, 256, 0, stream>>>(W1, B1, w1t);
  pack_w2_kernel<<<(D_OUT * K2 + 255) / 256, 256, 0, stream>>>(W2, B2, w2t);
  pack_a1_kernel<<<(64 * D_IN)  / 256, 256, 0, stream>>>(A1, a1t);
  pack_a2_kernel<<<(64 * D_HID) / 256, 256, 0, stream>>>(A2, a2t);

  // 256 M-rows per block (8 waves x 32 rows)
  const int mblocks = N_ROWS / 256;   // 64

  // LoRA-down 1 (WMMA): xa[:,512:576] = 2*bw ⊙ (x @ A1), K=512
  gemm_wmma_kernel<D_IN, K1, EPI_LORA, bf16><<<mblocks, 256, 0, stream>>>(
      xa, a1t, bw, xa + D_IN, K1, 1);

  // Stage 1: h = gelu(xa @ W1a + b1) -> ha[:, :2048] (bf16, stride 2112)
  gemm_wmma_kernel<K1, K1, EPI_GELU, bf16><<<mblocks * (D_HID / 64), 256, 0, stream>>>(
      xa, w1t, b1, ha, K2, D_HID / 64);

  // LoRA-down 2 (WMMA): ha[:,2048:2112] = 2*bw ⊙ (h @ A2), K=2048
  gemm_wmma_kernel<D_HID, K2, EPI_LORA, bf16><<<mblocks, 256, 0, stream>>>(
      ha, a2t, bw, ha + D_HID, K2, 1);

  // Stage 2: out = ha @ W2a + b2 (fp32)
  gemm_wmma_kernel<K2, K2, EPI_BIAS, float><<<mblocks * (D_OUT / 64), 256, 0, stream>>>(
      ha, w2t, b2, out, D_OUT, D_OUT / 64);
}
